// LatentCorrelationGCN_89773406421510
// MI455X (gfx1250) — matrix operand order repaired and checked by reading, not compiled, not run on hardware
//
#include <hip/hip_runtime.h>

#ifndef __has_builtin
#define __has_builtin(x) 0
#endif

#define BATCH 32
#define NNODE 1024
#define FDIM  256
#define KNEIGH 64
#define RTOT  (BATCH * NNODE)            // 32768 rows
#define WMAT  (FDIM * FDIM)              // 65536 elements per weight matrix

typedef __attribute__((ext_vector_type(16))) __bf16 v16bf;
typedef __attribute__((ext_vector_type(8)))  float  v8f;
typedef __attribute__((ext_vector_type(4)))  int    v4i;

union BFrag { v16bf v; unsigned u[8]; };
union AccF  { v8f v; float f[8]; };

__device__ inline unsigned f2bf(float f) {
  unsigned u = __float_as_uint(f);
  return (u + 0x7FFFu + ((u >> 16) & 1u)) >> 16;
}
__device__ inline unsigned pack2bf(float lo, float hi) {
#if __has_builtin(__builtin_amdgcn_cvt_pk_bf16_f32)
  typedef __attribute__((ext_vector_type(2))) __bf16 v2bf;
  union { v2bf v; unsigned u; } r;
  r.v = __builtin_amdgcn_cvt_pk_bf16_f32(lo, hi);
  return r.u;
#else
  return f2bf(lo) | (f2bf(hi) << 16);
#endif
}
__device__ inline unsigned fkey(float f) {          // monotone float->uint map
  unsigned u = __float_as_uint(f);
  return u ^ ((unsigned)((int)u >> 31) | 0x80000000u);
}
__device__ inline v8f wmma_bf16(const BFrag& a, const BFrag& b, v8f c) {
  return __builtin_amdgcn_wmma_f32_16x16x32_bf16(false, a.v, false, b.v, (short)0, c, false, false);
}
__device__ inline int   wsum_i(int x)  { for (int o = 16; o; o >>= 1) x += __shfl_xor(x, o, 32); return x; }
__device__ inline float wsum_f(float x){ for (int o = 16; o; o >>= 1) x += __shfl_xor(x, o, 32); return x; }
__device__ inline float wmax_f(float x){ for (int o = 16; o; o >>= 1) x = fmaxf(x, __shfl_xor(x, o, 32)); return x; }

// WMMA bf16 operand K-index maps (ISA 7.12.2, wave32)
__device__ inline int aK(int v, int h) { return (v < 4) ? (8*h + 2*v) : (16 + 8*h + 2*(v - 4)); }
__device__ inline int bK(int v, int h) { return (v < 4) ? (8*h + 2*v) : (16 + 8*h + 2*(v - 4)); }

// ---- gfx1250 async global->LDS copy (ASYNCcnt), with safe fallback ----------
#if __has_builtin(__builtin_amdgcn_global_load_async_to_lds_b128)
#define HAVE_ASYNC_LDS 1
__device__ inline void async_cp16(const void* g, void* l) {
  __builtin_amdgcn_global_load_async_to_lds_b128((v4i*)g, (v4i*)l, 0, 0);
}
__device__ inline void async_wait0() {
#if __has_builtin(__builtin_amdgcn_s_wait_asynccnt)
  __builtin_amdgcn_s_wait_asynccnt(0);
#else
  asm volatile("s_wait_asynccnt 0" ::: "memory");
#endif
}
#else
#define HAVE_ASYNC_LDS 0
#endif

// Copy a contiguous 16KB fp32 tile (16 rows x 256) into LDS staging (async if
// available), then pair-pack to bf16. 256 threads.
__device__ inline void stage_pack_tile(const float* __restrict__ src,
                                       float* __restrict__ stage,      // 16*256 LDS
                                       unsigned* __restrict__ packed,  // 16*128 LDS
                                       int tid) {
#if HAVE_ASYNC_LDS
  #pragma unroll
  for (int it = 0; it < 4; ++it) {
    const int off = it * 4096 + tid * 16;   // bytes
    async_cp16((const char*)src + off, (char*)stage + off);
  }
  async_wait0();
  __syncthreads();
  for (int i = tid; i < 16 * 128; i += 256)
    packed[i] = pack2bf(stage[2 * i], stage[2 * i + 1]);
#else
  (void)stage;
  for (int i = tid; i < 16 * 128; i += 256)
    packed[i] = pack2bf(src[2 * i], src[2 * i + 1]);
#endif
  __syncthreads();
}

// ------------------------------------------------ kernel 0: weights -> bf16 W^T
// wT[mat][o*256 + k] = bf16(W_mat[k*256 + o]); mats: wq,wk,wv,w1,w2
__global__ __launch_bounds__(256) void prep_kernel(
    const float* __restrict__ wq, const float* __restrict__ wk,
    const float* __restrict__ wv, const float* __restrict__ w1,
    const float* __restrict__ w2, unsigned short* __restrict__ wT)
{
  const int e = blockIdx.x * 256 + threadIdx.x;        // < 5*65536
  const int mat = e >> 16, rem = e & 65535;
  const int o = rem >> 8, k = rem & 255;
  const float* W = (mat == 0) ? wq : (mat == 1) ? wk :
                   (mat == 2) ? wv : (mat == 3) ? w1 : w2;
  wT[e] = (unsigned short)pack2bf(W[k * FDIM + o], 0.f);
}

// ---------------------------------------------------------------- kernel 1: QKV
__global__ __launch_bounds__(256) void qkv_kernel(
    const float* __restrict__ x, const unsigned* __restrict__ wTu,
    const float* __restrict__ bq, const float* __restrict__ bk,
    const float* __restrict__ bv,
    unsigned short* __restrict__ qkvb,   // [2][R][256] bf16 (q,k)
    unsigned short* __restrict__ vbT,    // [B][256][1024] bf16 (V^T)
    float* __restrict__ vf)              // [R][256] fp32
{
#if HAVE_ASYNC_LDS
  __shared__ float xf[16 * 256];
#else
  float* xf = nullptr;
#endif
  __shared__ unsigned xt[16 * 128];
  __shared__ float vt[8 * 272];          // per-wave 16x16 transpose pad-17
  const int r0  = blockIdx.x * 16;
  const int tid = threadIdx.x;
  stage_pack_tile(x + (size_t)r0 * FDIM, xf, xt, tid);

  const int wave = tid >> 5, lane = tid & 31;
  const int m = lane & 15, h = lane >> 4, n = lane & 15;
  const int bb = r0 >> 10, node0 = r0 & 1023;

  // A-fragments (x tile) shared by every output tile: load once.
  BFrag afr[8];
  #pragma unroll
  for (int kc = 0; kc < 8; ++kc)
    #pragma unroll
    for (int v = 0; v < 8; ++v)
      afr[kc].u[v] = xt[m * 128 + ((kc * 32 + aK(v, h)) >> 1)];

  // ---- Q and K: 32 tiles, branch-free stores --------------------------------
  for (int t = wave; t < 32; t += 8) {
    const int mat = t >> 4, col0 = (t & 15) * 16;
    const unsigned* Wt = wTu + (size_t)mat * (WMAT / 2) + (size_t)(col0 + n) * 128;
    BFrag bfr[8];
    #pragma unroll
    for (int kc = 0; kc < 8; ++kc)
      #pragma unroll
      for (int v = 0; v < 8; ++v) bfr[kc].u[v] = Wt[(kc * 32 + bK(v, h)) >> 1];
    AccF acc;
    const float bias = (mat == 0) ? bq[col0 + n] : bk[col0 + n];
    #pragma unroll
    for (int v = 0; v < 8; ++v) acc.f[v] = bias;
    #pragma unroll
    for (int kc = 0; kc < 8; ++kc) acc.v = wmma_bf16(afr[kc], bfr[kc], acc.v);
    unsigned short* dst = qkvb + (size_t)mat * ((size_t)RTOT * FDIM);
    #pragma unroll
    for (int v = 0; v < 8; ++v) {
      const int mm = v + 8 * h;
      dst[(size_t)(r0 + mm) * FDIM + col0 + n] = (unsigned short)pack2bf(acc.f[v], 0.f);
    }
  }

  // ---- V: 16 tiles; fp32 row-major + bf16 V^T via in-wave LDS transpose -----
  const unsigned* wvT = wTu + 2 * (size_t)(WMAT / 2);
  float* vtw = vt + wave * 272;
  for (int ct = wave; ct < 16; ct += 8) {
    const int col0 = ct * 16;
    const unsigned* Wt = wvT + (size_t)(col0 + n) * 128;
    BFrag bfr[8];
    #pragma unroll
    for (int kc = 0; kc < 8; ++kc)
      #pragma unroll
      for (int v = 0; v < 8; ++v) bfr[kc].u[v] = Wt[(kc * 32 + bK(v, h)) >> 1];
    AccF acc;
    const float bias = bv[col0 + n];
    #pragma unroll
    for (int v = 0; v < 8; ++v) acc.f[v] = bias;
    #pragma unroll
    for (int kc = 0; kc < 8; ++kc) acc.v = wmma_bf16(afr[kc], bfr[kc], acc.v);
    #pragma unroll
    for (int v = 0; v < 8; ++v) {
      const int mm = v + 8 * h;
      vf[(size_t)(r0 + mm) * FDIM + col0 + n] = acc.f[v];
      vtw[n * 17 + mm] = acc.f[v];                     // [col][row] stage
    }
    __builtin_amdgcn_wave_barrier();
    // lane (c=n, half h) emits 8 nodes of column col0+c as one b128 store
    const float* colp = vtw + n * 17 + h * 8;
    unsigned o0 = pack2bf(colp[0], colp[1]), o1 = pack2bf(colp[2], colp[3]);
    unsigned o2 = pack2bf(colp[4], colp[5]), o3 = pack2bf(colp[6], colp[7]);
    uint4* dst = (uint4*)(vbT + ((size_t)bb * FDIM + col0 + n) * NNODE + node0 + h * 8);
    *dst = make_uint4(o0, o1, o2, o3);
    __builtin_amdgcn_wave_barrier();
  }
}

// ------------------------------ kernel 2: scores -> top-k -> softmax -> attn*V
__global__ __launch_bounds__(256) void attn_kernel(
    const unsigned* __restrict__ qbu, const unsigned* __restrict__ kbu,
    const unsigned* __restrict__ vbTu,
    const float* __restrict__ sew,
    float* __restrict__ attn_out, float* __restrict__ msg)
{
  __shared__ float    sc[16 * NNODE];   // 64 KB fp32 scores
  __shared__ unsigned at[16 * 512];     // 32 KB bf16 attn, pair-packed
  const int bidx = blockIdx.x >> 6;
  const int q0   = (blockIdx.x & 63) * 16;
  const int tid  = threadIdx.x;
  const int wave = tid >> 5, lane = tid & 31;
  const int m = lane & 15, h = lane >> 4, n = lane & 15;
  const size_t rowbase = (size_t)bidx * NNODE;

  // Q-tile A-fragments: load once per wave, reuse for all 8 key tiles.
  BFrag afr[8];
  {
    const unsigned* qrow = qbu + (rowbase + q0 + m) * 128;
    #pragma unroll
    for (int kc = 0; kc < 8; ++kc)
      #pragma unroll
      for (int v = 0; v < 8; ++v)
        afr[kc].u[v] = qrow[(kc * 32 + aK(v, h)) >> 1];
  }

  // Phase 1: scores = Q K^T / 16 + static bias
  for (int jt = wave; jt < 64; jt += 8) {
    const int col0 = jt * 16;
    const unsigned* krow = kbu + (rowbase + col0 + n) * 128;
    BFrag bfr[8];
    #pragma unroll
    for (int kc = 0; kc < 8; ++kc)
      #pragma unroll
      for (int v = 0; v < 8; ++v) bfr[kc].u[v] = krow[(kc * 32 + bK(v, h)) >> 1];
    AccF acc;
    #pragma unroll
    for (int v = 0; v < 8; ++v) acc.f[v] = 0.f;
    #pragma unroll
    for (int kc = 0; kc < 8; ++kc) acc.v = wmma_bf16(afr[kc], bfr[kc], acc.v);
    #pragma unroll
    for (int v = 0; v < 8; ++v) {
      const int mm = v + 8 * h;
      sc[mm * NNODE + col0 + n] =
          acc.f[v] * 0.0625f + sew[(size_t)(q0 + mm) * NNODE + col0 + n];
    }
  }
  __syncthreads();

  // Phase 2+3: wave owns rows {2w, 2w+1}: exact 64th-largest via radix select,
  // then masked softmax (matches -1e9 masking semantics).
  for (int rr = 0; rr < 2; ++rr) {
    const int r = wave * 2 + rr;
    float val[32]; unsigned key[32];
    const float4* rowv = (const float4*)(sc + r * NNODE);
    float mx = -3.4e38f;
    #pragma unroll
    for (int j = 0; j < 8; ++j) {
      float4 q4 = rowv[lane * 8 + j];
      val[4*j+0] = q4.x; val[4*j+1] = q4.y; val[4*j+2] = q4.z; val[4*j+3] = q4.w;
    }
    #pragma unroll
    for (int i = 0; i < 32; ++i) { key[i] = fkey(val[i]); mx = fmaxf(mx, val[i]); }
    mx = wmax_f(mx);
    unsigned prefix = 0;
    for (int bit = 31; bit >= 0; --bit) {
      const unsigned cand = prefix | (1u << bit);
      int cnt = 0;
      #pragma unroll
      for (int i = 0; i < 32; ++i) cnt += (key[i] >= cand) ? 1 : 0;
      cnt = wsum_i(cnt);
      if (cnt >= KNEIGH) prefix = cand;
    }
    float s = 0.f;
    #pragma unroll
    for (int i = 0; i < 32; ++i) {
      const float e = (key[i] >= prefix) ? __expf(val[i] - mx) : 0.f;
      val[i] = e; s += e;
    }
    s = wsum_f(s);
    const float inv = 1.f / s;
    float* arow = attn_out + (rowbase + q0 + r) * (size_t)NNODE + lane * 32;
    #pragma unroll
    for (int j = 0; j < 8; ++j)
      ((float4*)arow)[j] = make_float4(val[4*j]*inv, val[4*j+1]*inv,
                                       val[4*j+2]*inv, val[4*j+3]*inv);
    #pragma unroll
    for (int p = 0; p < 16; ++p)
      at[r * 512 + lane * 16 + p] = pack2bf(val[2*p] * inv, val[2*p+1] * inv);
  }
  __syncthreads();

  // Phase 4: messages = attn (16x1024) x V^T, group-of-4 software pipelining
  for (int ct = wave; ct < 16; ct += 8) {
    const int col0 = ct * 16;
    const unsigned* vcol = vbTu + ((size_t)bidx * FDIM + col0 + n) * (NNODE / 2);
    AccF acc;
    #pragma unroll
    for (int v = 0; v < 8; ++v) acc.f[v] = 0.f;
    for (int g = 0; g < 8; ++g) {
      BFrag ag[4], bg[4];
      #pragma unroll
      for (int q = 0; q < 4; ++q) {
        const int k0 = (g * 4 + q) * 32;
        #pragma unroll
        for (int v = 0; v < 8; ++v) {
          ag[q].u[v] = at[m * 512 + ((k0 + aK(v, h)) >> 1)];
          bg[q].u[v] = vcol[(k0 + bK(v, h)) >> 1];
        }
      }
      #pragma unroll
      for (int q = 0; q < 4; ++q) acc.v = wmma_bf16(ag[q], bg[q], acc.v);
    }
    #pragma unroll
    for (int v = 0; v < 8; ++v) {
      const int mm = v + 8 * h;
      msg[(rowbase + q0 + mm) * (size_t)FDIM + col0 + n] = acc.f[v];
    }
  }
}

// ---------------------------------------------------------------- kernel 3: MLP
__global__ __launch_bounds__(256) void mlp_kernel(
    const float* __restrict__ msg, const unsigned* __restrict__ wTu,
    const float* __restrict__ b1,
    const float* __restrict__ ln_g, const float* __restrict__ ln_b,
    const float* __restrict__ b2,
    const float* __restrict__ vf, float* __restrict__ out)
{
#if HAVE_ASYNC_LDS
  __shared__ float mf[16 * 256];
#else
  float* mf = nullptr;
#endif
  __shared__ unsigned mt[16 * 128];
  __shared__ float    ht[16 * 256];
  __shared__ unsigned hb[16 * 128];
  const int r0  = blockIdx.x * 16;
  const int tid = threadIdx.x;
  stage_pack_tile(msg + (size_t)r0 * FDIM, mf, mt, tid);

  const int wave = tid >> 5, lane = tid & 31;
  const int m = lane & 15, h = lane >> 4, n = lane & 15;
  const unsigned* w1t = wTu + 3 * (size_t)(WMAT / 2);
  const unsigned* w2t = wTu + 4 * (size_t)(WMAT / 2);

  // h = msg @ w1 + b1
  {
    BFrag afr[8];
    #pragma unroll
    for (int kc = 0; kc < 8; ++kc)
      #pragma unroll
      for (int v = 0; v < 8; ++v)
        afr[kc].u[v] = mt[m * 128 + ((kc * 32 + aK(v, h)) >> 1)];
    for (int ct = wave; ct < 16; ct += 8) {
      const int col0 = ct * 16;
      const unsigned* Wt = w1t + (size_t)(col0 + n) * 128;
      BFrag bfr[8];
      #pragma unroll
      for (int kc = 0; kc < 8; ++kc)
        #pragma unroll
        for (int v = 0; v < 8; ++v) bfr[kc].u[v] = Wt[(kc * 32 + bK(v, h)) >> 1];
      AccF acc;
      const float bias = b1[col0 + n];
      #pragma unroll
      for (int v = 0; v < 8; ++v) acc.f[v] = bias;
      #pragma unroll
      for (int kc = 0; kc < 8; ++kc) acc.v = wmma_bf16(afr[kc], bfr[kc], acc.v);
      #pragma unroll
      for (int v = 0; v < 8; ++v) ht[(v + 8 * h) * 256 + col0 + n] = acc.f[v];
    }
  }
  __syncthreads();

  // LayerNorm + ReLU (fp32); wave owns rows {2w, 2w+1}, lane owns 8 cols
  for (int rr = 0; rr < 2; ++rr) {
    const int r = wave * 2 + rr;
    float x8[8];
    const float4* rp = (const float4*)(ht + r * 256);
    float4 a4 = rp[lane * 2], b4 = rp[lane * 2 + 1];
    x8[0]=a4.x; x8[1]=a4.y; x8[2]=a4.z; x8[3]=a4.w;
    x8[4]=b4.x; x8[5]=b4.y; x8[6]=b4.z; x8[7]=b4.w;
    float s = 0.f, sq = 0.f;
    #pragma unroll
    for (int j = 0; j < 8; ++j) { s += x8[j]; sq += x8[j] * x8[j]; }
    s = wsum_f(s); sq = wsum_f(sq);
    const float mu   = s * (1.f / 256.f);
    const float var  = sq * (1.f / 256.f) - mu * mu;
    const float rstd = rsqrtf(var + 1e-5f);
    const int c0 = lane * 8;
    #pragma unroll
    for (int j = 0; j < 8; ++j) {
      const float g = (x8[j] - mu) * rstd * ln_g[c0 + j] + ln_b[c0 + j];
      x8[j] = fmaxf(g, 0.f);
    }
    #pragma unroll
    for (int p = 0; p < 4; ++p)
      hb[r * 128 + lane * 4 + p] = pack2bf(x8[2*p], x8[2*p+1]);
  }
  __syncthreads();

  // out = relu_h @ w2 + b2 + v
  {
    BFrag afr[8];
    #pragma unroll
    for (int kc = 0; kc < 8; ++kc)
      #pragma unroll
      for (int v = 0; v < 8; ++v)
        afr[kc].u[v] = hb[m * 128 + ((kc * 32 + aK(v, h)) >> 1)];
    for (int ct = wave; ct < 16; ct += 8) {
      const int col0 = ct * 16;
      const unsigned* Wt = w2t + (size_t)(col0 + n) * 128;
      BFrag bfr[8];
      #pragma unroll
      for (int kc = 0; kc < 8; ++kc)
        #pragma unroll
        for (int v = 0; v < 8; ++v) bfr[kc].u[v] = Wt[(kc * 32 + bK(v, h)) >> 1];
      AccF acc;
      const float bias = b2[col0 + n];
      #pragma unroll
      for (int v = 0; v < 8; ++v) acc.f[v] = bias;
      #pragma unroll
      for (int kc = 0; kc < 8; ++kc) acc.v = wmma_bf16(afr[kc], bfr[kc], acc.v);
      #pragma unroll
      for (int v = 0; v < 8; ++v) {
        const int mm = v + 8 * h;
        const size_t idx = (size_t)(r0 + mm) * FDIM + col0 + n;
        out[idx] = acc.f[v] + vf[idx];
      }
    }
  }
}

// ---------------------------------------------------------------- launch
extern "C" void kernel_launch(void* const* d_in, const int* in_sizes, int n_in,
                              void* d_out, int out_size, void* d_ws, size_t ws_size,
                              hipStream_t stream) {
  (void)in_sizes; (void)n_in; (void)out_size; (void)ws_size;
  const float* x    = (const float*)d_in[0];
  const float* wq   = (const float*)d_in[1];
  const float* bq   = (const float*)d_in[2];
  const float* wk   = (const float*)d_in[3];
  const float* bk   = (const float*)d_in[4];
  const float* wv   = (const float*)d_in[5];
  const float* bv   = (const float*)d_in[6];
  const float* sew  = (const float*)d_in[7];
  const float* w1   = (const float*)d_in[8];
  const float* b1   = (const float*)d_in[9];
  const float* ln_g = (const float*)d_in[10];
  const float* ln_b = (const float*)d_in[11];
  const float* w2   = (const float*)d_in[12];
  const float* b2   = (const float*)d_in[13];

  float* out_ptr  = (float*)d_out;                              // [B,N,F]
  float* attn_ptr = out_ptr + (size_t)RTOT * FDIM;              // [B,N,N]

  char* ws = (char*)d_ws;
  const size_t MB16 = (size_t)RTOT * FDIM * 2;   // one bf16 [R,256] matrix
  unsigned short* qkvb = (unsigned short*)(ws);                 // q,k: 2 x 16MB
  unsigned short* vbT  = (unsigned short*)(ws + 2 * MB16);      // V^T: 16MB
  float*          vf   = (float*)(ws + 3 * MB16);               // 32MB
  float*          mg   = (float*)(ws + 3 * MB16 + (size_t)RTOT * FDIM * 4);
  unsigned short* wT   = (unsigned short*)(ws + 3 * MB16 + 2 * (size_t)RTOT * FDIM * 4);

  const int tiles = RTOT / 16;   // 2048
  prep_kernel<<<(5 * WMAT) / 256, 256, 0, stream>>>(wq, wk, wv, w1, w2, wT);
  qkv_kernel<<<tiles, 256, 0, stream>>>(x, (const unsigned*)wT, bq, bk, bv,
                                        qkvb, vbT, vf);
  attn_kernel<<<tiles, 256, 0, stream>>>((const unsigned*)qkvb,
                                         (const unsigned*)(qkvb + RTOT * (size_t)FDIM),
                                         (const unsigned*)vbT, sew, attn_ptr, mg);
  mlp_kernel<<<tiles, 256, 0, stream>>>(mg, (const unsigned*)wT, b1, ln_g, ln_b,
                                        b2, vf, out_ptr);
}
